// SDFNetwork_29953101922452
// MI455X (gfx1250) — matrix operand order repaired and checked by reading, not compiled, not run on hardware
//
#include <hip/hip_runtime.h>
#include <hip/hip_bf16.h>
#include <math.h>

// ---------------- types ----------------
typedef _Float16 h16;
typedef __attribute__((ext_vector_type(16))) _Float16 v16h;
typedef __attribute__((ext_vector_type(8)))  float    v8f;

// ---------------- hashgrid config ----------------
#define NLEV   16
#define TBL_SZ (1u << 19)          // T = 2^19 entries per level, F=2 floats each
#define BLK    256

// scale_l = 16 * 1.3819^l - 1  (precomputed, matches tcnn/jnp float32)
__constant__ float c_scales[NLEV] = {
    15.0f,        21.1104f,     29.5543618f,  41.2230724f,  57.3480638f,
    79.6311894f,  110.424241f,  152.977158f,  211.781035f,  293.042112f,
    405.336795f,  560.516817f,  774.960089f,  1071.29925f,  1480.81033f,
    2046.71370f
};
// res_l = ceil(scale_l)+1; levels 0..4 are dense (res^3 <= T), 5..15 hashed
__constant__ uint32_t c_res[5] = {16u, 23u, 31u, 43u, 59u};

#define PRIME_Y 2654435761u
#define PRIME_Z 805459861u

// LDS strides (in halves); odd dword stride => conflict-free across 64 LDS banks
#define XS 34   // 32 features + 2 pad
#define HS 66   // 64 hidden  + 2 pad

// ---------------- WMMA helpers ----------------
union AFrag { v16h v; h16 h[16]; uint32_t u[8]; };
union BFrag { v16h v; h16 h[16]; };
union PK2   { h16 h[2]; uint32_t u; };

// A-matrix 16x32 f16 fragment from LDS tile (row 0 at `base`, stride in halves).
// ISA layout: lane = half*16 + m; VGPR v<4: K = half*8 + 2v(,+1); v>=4: K = 16+half*8+2(v-4).
__device__ __forceinline__ v16h lds_loadA(const h16* base, int strideH, int kbase, int lane) {
    AFrag a;
    const int hs = lane >> 4, m = lane & 15;
    const uint32_t* r32 = (const uint32_t*)(base + m * strideH + kbase);
    const int o = hs * 4;
#pragma unroll
    for (int v = 0; v < 4; ++v) a.u[v]     = r32[o + v];
#pragma unroll
    for (int v = 0; v < 4; ++v) a.u[4 + v] = r32[8 + o + v];
    return a.v;
}

// B-matrix 32x16 f16 fragment from global f32 row-major W[K][ncols].
// ISA layout: element i<8: K = half*8 + i; i>=8: K = 16 + half*8 + (i-8); N = lane%16.
__device__ __forceinline__ v16h g_loadB(const float* __restrict__ W, int ncols,
                                        int kb, int nb, int lane) {
    BFrag b;
    const int n = lane & 15;
    const int kr = (lane >> 4) * 8;
    const float* p = W + (size_t)(kb + kr) * ncols + nb * 16 + n;
#pragma unroll
    for (int i = 0; i < 16; ++i) b.h[i] = (h16)p[(i + (i >= 8 ? 8 : 0)) * ncols];
    return b.v;
}

__device__ __forceinline__ v8f wmma_f16(v16h a, v16h b, v8f c) {
    return __builtin_amdgcn_wmma_f32_16x16x32_f16(
        /*neg_a=*/false, a, /*neg_b=*/false, b,
        /*c_mod=*/(short)0, c, /*reuse_a=*/false, /*reuse_b=*/false);
}

// softplus(100*v)/100, branchless, hardware exp/log (v_exp_f32/v_log_f32)
__device__ __forceinline__ float softplus100(float v) {
    const float z  = 100.0f * v;
    const float zc = fminf(z, 20.0f);
    const float sp = 0.01f * __logf(1.0f + __expf(zc));
    return (z > 20.0f) ? v : sp;      // v_cndmask, no EXEC change
}

// ---------------- fused kernel ----------------
__global__ __launch_bounds__(BLK)
void hashgrid_mlp_kernel(const float* __restrict__ inputs,   // [N,3]
                         const float* __restrict__ tables,   // [L,T,2]
                         const float* __restrict__ W1, const float* __restrict__ b1,
                         const float* __restrict__ W2, const float* __restrict__ b2,
                         const float* __restrict__ W3, const float* __restrict__ b3,
                         float* __restrict__ out)             // [N,16]
{
    __shared__ h16 sX[BLK * XS];   // encoded features, f16
    __shared__ h16 sH[BLK * HS];   // hidden activations, f16 (h1 -> h2 in place)

    const int tid = threadIdx.x;
    const int pt0 = blockIdx.x * BLK;

    // ================= hash-grid encode (L2-resident gathers) =================
    {
        const int pt = pt0 + tid;
        const float x = (inputs[(size_t)pt * 3 + 0] + 3.0f) * (1.0f / 6.0f);
        const float y = (inputs[(size_t)pt * 3 + 1] + 3.0f) * (1.0f / 6.0f);
        const float z = (inputs[(size_t)pt * 3 + 2] + 3.0f) * (1.0f / 6.0f);
        uint32_t* sX32 = (uint32_t*)sX;

#pragma unroll
        for (int l = 0; l < NLEV; ++l) {
            const float s = c_scales[l];
            const float px = fmaf(x, s, 0.5f);
            const float py = fmaf(y, s, 0.5f);
            const float pz = fmaf(z, s, 0.5f);
            const float fx = floorf(px), fy = floorf(py), fz = floorf(pz);
            const float wx = px - fx, wy = py - fy, wz = pz - fz;
            const uint32_t cx = (uint32_t)(int)fx;
            const uint32_t cy = (uint32_t)(int)fy;
            const uint32_t cz = (uint32_t)(int)fz;
            const float2* tab = (const float2*)(tables + (size_t)l * TBL_SZ * 2u);

            float f0 = 0.0f, f1 = 0.0f;
#pragma unroll
            for (int c = 0; c < 8; ++c) {
                const uint32_t ux = cx + (uint32_t)(c & 1);
                const uint32_t uy = cy + (uint32_t)((c >> 1) & 1);
                const uint32_t uz = cz + (uint32_t)((c >> 2) & 1);
                uint32_t idx;
                if (l < 5) {                       // dense tight indexing
                    const uint32_t r = c_res[l];
                    idx = ux + uy * r + uz * r * r;
                } else {                           // spatial hash
                    idx = ux ^ (uy * PRIME_Y) ^ (uz * PRIME_Z);
                }
                idx &= (TBL_SZ - 1u);
                const float2 t = tab[idx];
                const float w = ((c & 1) ? wx : 1.0f - wx) *
                                (((c >> 1) & 1) ? wy : 1.0f - wy) *
                                (((c >> 2) & 1) ? wz : 1.0f - wz);
                f0 = fmaf(t.x, w, f0);
                f1 = fmaf(t.y, w, f1);
            }
            PK2 pk; pk.h[0] = (h16)f0; pk.h[1] = (h16)f1;
            sX32[tid * (XS / 2) + l] = pk.u;       // features [2l, 2l+1] packed
        }
    }
    __syncthreads();

    // ============ fused MLP via WMMA (each wave owns 32 rows = 2 M-tiles) =====
    // Both M-tiles are processed per weight fragment so each B is loaded once.
    const int wave = tid >> 5;
    const int lane = tid & 31;
    const int hs   = lane >> 4;
    const int n    = lane & 15;
    const int m0   = wave * 32;        // M-tile 0 base row
    const int m1   = m0 + 16;          // M-tile 1 base row

    // ---- GEMM1: X[32r,32] @ W1[32,64] + b1, softplus -> sH (f16) ----
    {
        const v16h a0 = lds_loadA(sX + m0 * XS, XS, 0, lane);
        const v16h a1 = lds_loadA(sX + m1 * XS, XS, 0, lane);
#pragma unroll
        for (int nb = 0; nb < 4; ++nb) {
            const v16h b = g_loadB(W1, 64, 0, nb, lane);
            v8f c0 = {}, c1 = {};
            c0 = wmma_f16(a0, b, c0);
            c1 = wmma_f16(a1, b, c1);
            const float bias = b1[nb * 16 + n];
#pragma unroll
            for (int i = 0; i < 8; ++i) {
                const int col = nb * 16 + n;
                sH[(m0 + i + 8 * hs) * HS + col] = (h16)softplus100(c0[i] + bias);
                sH[(m1 + i + 8 * hs) * HS + col] = (h16)softplus100(c1[i] + bias);
            }
        }
    }
    // wave-private rows; per-wave LDS in-order + explicit dscnt wait suffices
    asm volatile("s_wait_dscnt 0" ::: "memory");

    // ---- GEMM2: H1[32r,64] @ W2[64,64] + b2 (no act), in-place in sH ----
    {
        // pre-load all A K-slices so in-place overwrite is hazard-free
        const v16h a00 = lds_loadA(sH + m0 * HS, HS, 0,  lane);
        const v16h a01 = lds_loadA(sH + m0 * HS, HS, 32, lane);
        const v16h a10 = lds_loadA(sH + m1 * HS, HS, 0,  lane);
        const v16h a11 = lds_loadA(sH + m1 * HS, HS, 32, lane);
        asm volatile("" ::: "memory");   // keep loads above the stores below
#pragma unroll
        for (int nb = 0; nb < 4; ++nb) {
            const v16h b0 = g_loadB(W2, 64, 0,  nb, lane);
            const v16h b1f = g_loadB(W2, 64, 32, nb, lane);
            v8f c0 = {}, c1 = {};
            c0 = wmma_f16(a00, b0,  c0);
            c0 = wmma_f16(a01, b1f, c0);
            c1 = wmma_f16(a10, b0,  c1);
            c1 = wmma_f16(a11, b1f, c1);
            const float bias = b2[nb * 16 + n];
#pragma unroll
            for (int i = 0; i < 8; ++i) {
                const int col = nb * 16 + n;
                sH[(m0 + i + 8 * hs) * HS + col] = (h16)(c0[i] + bias);
                sH[(m1 + i + 8 * hs) * HS + col] = (h16)(c1[i] + bias);
            }
        }
    }
    asm volatile("s_wait_dscnt 0" ::: "memory");

    // ---- GEMM3: H2[32r,64] @ W3[64,16] + b3 -> global out ----
    {
        const v16h a00 = lds_loadA(sH + m0 * HS, HS, 0,  lane);
        const v16h a01 = lds_loadA(sH + m0 * HS, HS, 32, lane);
        const v16h a10 = lds_loadA(sH + m1 * HS, HS, 0,  lane);
        const v16h a11 = lds_loadA(sH + m1 * HS, HS, 32, lane);
        const v16h b0 = g_loadB(W3, 16, 0,  0, lane);
        const v16h b1f = g_loadB(W3, 16, 32, 0, lane);
        v8f c0 = {}, c1 = {};
        c0 = wmma_f16(a00, b0,  c0);
        c0 = wmma_f16(a01, b1f, c0);
        c1 = wmma_f16(a10, b0,  c1);
        c1 = wmma_f16(a11, b1f, c1);
        const float bias = b3[n];
#pragma unroll
        for (int i = 0; i < 8; ++i) {
            out[(size_t)(pt0 + m0 + i + 8 * hs) * 16 + n] = c0[i] + bias;
            out[(size_t)(pt0 + m1 + i + 8 * hs) * 16 + n] = c1[i] + bias;
        }
    }
}

// ---------------- launch ----------------
extern "C" void kernel_launch(void* const* d_in, const int* in_sizes, int n_in,
                              void* d_out, int out_size, void* d_ws, size_t ws_size,
                              hipStream_t stream) {
    const float* inputs = (const float*)d_in[0];   // [N,3]
    const float* tables = (const float*)d_in[1];   // [16, 2^19, 2]
    const float* W1     = (const float*)d_in[2];   // [32,64]
    const float* b1     = (const float*)d_in[3];   // [64]
    const float* W2     = (const float*)d_in[4];   // [64,64]
    const float* b2     = (const float*)d_in[5];   // [64]
    const float* W3     = (const float*)d_in[6];   // [64,16]
    const float* b3     = (const float*)d_in[7];   // [16]
    float* out          = (float*)d_out;           // [N,16]

    const int npts = in_sizes[0] / 3;
    const int grid = (npts + BLK - 1) / BLK;
    hashgrid_mlp_kernel<<<grid, BLK, 0, stream>>>(inputs, tables, W1, b1, W2, b2,
                                                  W3, b3, out);
}
